// LongTermMemory_85126251807430
// MI455X (gfx1250) — matrix operand order repaired and checked by reading, not compiled, not run on hardware
//
#include <hip/hip_runtime.h>
#include <hip/hip_bf16.h>
#include <math.h>

#define KDIM        256
#define MROWS       100000
#define BROWS       2048
#define VDIM        256
#define MEM_STRIDE  512     // memory rows are [256 key | 256 value] floats

// ---- main-kernel tiling -----------------------------------------------------
#define TM            32    // key rows per LDS stage (2 tiles of 16)
#define TILES_STAGE   2
#define LDS_STRIDE    264   // bf16 elems per LDS row: 256 + 8 pad (16B) -> bank rotation
#define NSTAGES       3125  // 100000 / 32
#define GRIDX         125   // stages strided: 3125/125 = 25 per block
#define CHUNKS_THR    4     // (TM rows * 32 chunks of 16B) / 256 threads

typedef __attribute__((ext_vector_type(16))) __bf16 v16bf;
typedef __attribute__((ext_vector_type(8)))  __bf16 v8bf;
typedef __attribute__((ext_vector_type(8)))  float  v8f;

// --- Pass 1: convert f32 rows -> bf16 workspace, compute 1/||row|| ----------
__global__ void prep_rows(const float* __restrict__ src, int srcStride, int nrows,
                          __bf16* __restrict__ dst, float* __restrict__ invn)
{
    const int lane = threadIdx.x & 31;
    const int wave = threadIdx.x >> 5;
    const int row  = blockIdx.x * 8 + wave;
    if (row >= nrows) return;

    const float* s = src + (size_t)row * srcStride + lane * 8;
    float ss = 0.f;
    v8bf b;
#pragma unroll
    for (int i = 0; i < 8; ++i) {
        float v = s[i];
        ss += v * v;
        b[i] = (__bf16)v;
    }
    *(v8bf*)(dst + (size_t)row * KDIM + lane * 8) = b;

#pragma unroll
    for (int off = 16; off > 0; off >>= 1)
        ss += __shfl_xor(ss, off, 32);
    if (lane == 0)
        invn[row] = 1.0f / fmaxf(sqrtf(ss), 1e-8f);
}

__global__ void init_best(unsigned long long* best)
{
    int i = blockIdx.x * blockDim.x + threadIdx.x;
    if (i < BROWS) best[i] = 0ull;
}

// --- async global->LDS copy of one 32x256 bf16 key stage --------------------
__device__ __forceinline__ void issue_stage_copy(const __bf16* __restrict__ keysBf,
                                                 int stage, __bf16* buf, int tid)
{
    const size_t gbase = (size_t)stage * TM * KDIM;
#pragma unroll
    for (int j = 0; j < CHUNKS_THR; ++j) {
        const int chunk = tid + j * 256;
        const int row   = chunk >> 5;   // 32 x 16B chunks per 512B key row
        const int col   = chunk & 31;
        const __bf16* g = keysBf + gbase + (size_t)row * KDIM + col * 8;
        unsigned ldsOff = (unsigned)(uintptr_t)(buf + row * LDS_STRIDE + col * 8);
        asm volatile("global_load_async_to_lds_b128 %0, %1, off"
                     :: "v"(ldsOff), "v"((unsigned long long)(uintptr_t)g)
                     : "memory");
    }
}

#define WAIT_ASYNC_PREV() asm volatile("s_wait_asynccnt 0x4" ::: "memory")
#define WAIT_ASYNC_ALL()  asm volatile("s_wait_asynccnt 0x0" ::: "memory")

// sched_group_barrier masks
#define SGB_WMMA    0x008
#define SGB_DSREAD  0x100

// --- Pass 2: fused bf16 WMMA GEMM + per-query running argmax ----------------
__global__ void __launch_bounds__(256)
sim_argmax(const __bf16* __restrict__ keysBf,
           const __bf16* __restrict__ qBf,
           const float*  __restrict__ invKn,
           unsigned long long* __restrict__ best)
{
    __shared__ __bf16 smem[2][TM * LDS_STRIDE];   // 2 x 16.5 KB, double buffered

    const int tid  = threadIdx.x;
    const int lane = tid & 31;
    const int wave = tid >> 5;
    const int hi16 = lane >> 4;
    const int l15  = lane & 15;

    const int qbase = blockIdx.y * 128 + wave * 16;

    // Kick off stage-0 DMA before touching queries (overlap with preload).
    const int s0 = blockIdx.x;
    issue_stage_copy(keysBf, s0, smem[0], tid);

    // Preload all 8 query (B-matrix) fragments.
    v16bf q[8];
    const __bf16* qrow = qBf + (size_t)(qbase + l15) * KDIM + hi16 * 16;
#pragma unroll
    for (int kk = 0; kk < 8; ++kk)
        q[kk] = *(const v16bf*)(qrow + kk * 32);

    float    bestV = -3.0e38f;
    unsigned bestI = 0;

    int bufSel = 0;
    for (int s = s0; s < NSTAGES; s += GRIDX) {
        const int nxt = s + GRIDX;
        if (nxt < NSTAGES) {                       // uniform branch
            issue_stage_copy(keysBf, nxt, smem[bufSel ^ 1], tid);
            WAIT_ASYNC_PREV();                     // stage s landed (in-order)
        } else {
            WAIT_ASYNC_ALL();
        }
        __syncthreads();                           // stage s visible to all waves

        const __bf16* buf = smem[bufSel];
        const int sBase = s * TM;
#pragma unroll
        for (int tt = 0; tt < TILES_STAGE; ++tt) {
            const __bf16* arow = buf + (tt * 16 + l15) * LDS_STRIDE + hi16 * 16;
            const int mbase = sBase + tt * 16;
            v8f ik = *(const v8f*)(invKn + mbase + hi16 * 8);

            v16bf a[8];
#pragma unroll
            for (int kk = 0; kk < 8; ++kk) {
                v8bf lo = *(const v8bf*)(arow + kk * 32);
                v8bf hh = *(const v8bf*)(arow + kk * 32 + 8);
                a[kk] = __builtin_shufflevector(lo, hh,
                            0,1,2,3,4,5,6,7,8,9,10,11,12,13,14,15);
            }

            v8f c = {};
#pragma unroll
            for (int kk = 0; kk < 8; ++kk)
                c = __builtin_amdgcn_wmma_f32_16x16x32_bf16(
                        false, a[kk], false, q[kk], (short)0, c, false, false);

#pragma unroll
            for (int j = 0; j < 8; ++j) {
                float sim = c[j] * ik[j];
                unsigned idx = (unsigned)(mbase + hi16 * 8 + j);
                if (sim > bestV) { bestV = sim; bestI = idx; }
            }
        }

#if __has_builtin(__builtin_amdgcn_sched_group_barrier)
        // Pin the per-stage pipeline: 32 DS reads + 16 WMMAs, 2-fragment
        // lookahead so every WMMA issues with operands resident while the
        // next fragment's loads are in flight (partial dscnt waits).
        __builtin_amdgcn_sched_group_barrier(SGB_DSREAD, 4, 0);   // frags 0,1
#pragma unroll
        for (int i = 0; i < 14; ++i) {
            __builtin_amdgcn_sched_group_barrier(SGB_WMMA,   1, 0);
            __builtin_amdgcn_sched_group_barrier(SGB_DSREAD, 2, 0);
        }
        __builtin_amdgcn_sched_group_barrier(SGB_WMMA, 2, 0);     // drain
#endif

        __syncthreads();                           // done reading before reuse
        bufSel ^= 1;
    }

    // Merge the two row-halves (lane n <-> n+16 hold the same query).
    float    oV = __shfl_xor(bestV, 16, 32);
    unsigned oI = (unsigned)__shfl_xor((int)bestI, 16, 32);
    if (oV > bestV || (oV == bestV && oI > bestI)) { bestV = oV; bestI = oI; }

    if (lane < 16) {
        unsigned u = __float_as_uint(bestV);
        u ^= (u & 0x80000000u) ? 0xFFFFFFFFu : 0x80000000u;
        unsigned long long packed = ((unsigned long long)u << 32) | bestI;
        atomicMax(best + qbase + l15, packed);
    }
}

// --- Pass 3: gather winning value rows --------------------------------------
__global__ void gather_values(const unsigned long long* __restrict__ best,
                              const float* __restrict__ memory,
                              float* __restrict__ out)
{
    const int b = blockIdx.x;
    const unsigned idx = (unsigned)(best[b] & 0xFFFFFFFFull);
    out[(size_t)b * VDIM + threadIdx.x] =
        memory[(size_t)idx * MEM_STRIDE + KDIM + threadIdx.x];
}

extern "C" void kernel_launch(void* const* d_in, const int* in_sizes, int n_in,
                              void* d_out, int out_size, void* d_ws, size_t ws_size,
                              hipStream_t stream)
{
    const float* query  = (const float*)d_in[0];   // (2048, 256)
    const float* memory = (const float*)d_in[1];   // (100000, 512)
    float* out = (float*)d_out;                    // (2048, 256)

    char* ws = (char*)d_ws;
    size_t off = 0;
    auto take = [&](size_t bytes) -> char* {
        char* p = ws + off;
        off = (off + bytes + 255) & ~(size_t)255;
        return p;
    };
    __bf16* keysBf = (__bf16*)take((size_t)MROWS * KDIM * sizeof(__bf16)); // 51.2 MB
    __bf16* qBf    = (__bf16*)take((size_t)BROWS * KDIM * sizeof(__bf16)); //  1.0 MB
    float*  invKn  = (float*) take((size_t)MROWS * sizeof(float));
    float*  invQn  = (float*) take((size_t)BROWS * sizeof(float));         // unused (argmax-invariant)
    unsigned long long* best = (unsigned long long*)take((size_t)BROWS * 8);

    prep_rows<<<MROWS / 8, 256, 0, stream>>>(memory, MEM_STRIDE, MROWS, keysBf, invKn);
    prep_rows<<<BROWS / 8, 256, 0, stream>>>(query,  KDIM,       BROWS, qBf,    invQn);
    init_best<<<(BROWS + 255) / 256, 256, 0, stream>>>(best);

    dim3 grid(GRIDX, BROWS / 128);   // 125 stage-strides x 16 query groups
    sim_argmax<<<grid, 256, 0, stream>>>(keysBf, qBf, invKn, best);

    gather_values<<<BROWS, VDIM, 0, stream>>>(best, memory, out);
}
